// XASStructure_41841571397765
// MI455X (gfx1250) — matrix operand order repaired and checked by reading, not compiled, not run on hardware
//
#include <hip/hip_runtime.h>
#include <cstdint>
#include <cstddef>

#define Hd   256
#define Nn   20000
#define Ee   640000
#define Gg   16
#define AD   118
#define ADP  128          // AD padded to multiple of 32
#define OD   100
#define Ll   2

typedef __attribute__((ext_vector_type(16))) __bf16          v16bf;
typedef __attribute__((ext_vector_type(8)))  float           v8f;
typedef __attribute__((ext_vector_type(16))) unsigned short  v16u;
typedef __attribute__((ext_vector_type(8)))  unsigned int    v8u;
typedef __attribute__((ext_vector_type(4)))  unsigned int    v4u;

__device__ __forceinline__ unsigned short f2bf(float f) {
  unsigned u = __builtin_bit_cast(unsigned, f);
  u += 0x7FFFu + ((u >> 16) & 1u);          // round-to-nearest-even
  return (unsigned short)(u >> 16);
}

// ---------------------------------------------------------------------------
// Pack fp32 weights W[K x 256] (row stride ldw) into bf16 WMMA B-operand
// tiles. Tile (kt,nt): 32x16, flat id kt*16+nt, 512 bf16.  In-tile: lane L
// holds column N = nt*16 + L%16; K = kt*32 + (s/8)*16 + (L/16)*8 + s%8.
// ---------------------------------------------------------------------------
__global__ __launch_bounds__(256) void pack_b_kernel(
    const float* __restrict__ W, int K, int ldw,
    unsigned short* __restrict__ out, int KT) {
  int t = blockIdx.x * 256 + threadIdx.x;
  if (t >= KT * 16 * 512) return;
  int tile = t >> 9, slot = t & 511;
  int lane = slot >> 4, s = slot & 15;
  int kt = tile >> 4, nt = tile & 15;
  int k = kt * 32 + ((s >> 3) << 4) + ((lane >> 4) << 3) + (s & 7);
  int n = nt * 16 + (lane & 15);
  float v = (k < K) ? W[(size_t)k * ldw + n] : 0.f;
  out[t] = f2bf(v);
}

// atomic_num (N x 118 fp32) -> bf16, row stride padded to 128 with zeros
__global__ __launch_bounds__(256) void cvt_atom_kernel(
    const float* __restrict__ in, unsigned short* __restrict__ out, int total) {
  int i = blockIdx.x * 256 + threadIdx.x;
  if (i >= total) return;
  int n = i >> 7, j = i & (ADP - 1);
  out[i] = (j < AD) ? f2bf(in[(size_t)n * AD + j]) : (unsigned short)0;
}

// ---------------------------------------------------------------------------
// C[M x 256] (+)= A_bf16[M x KT*32] @ Bpacked + bias.
// Requirements: M % 16 == 0 (all calls use M = 20000), A zero-padded in K.
// Block: 256 thr = 8 wave32; wave w owns output col tiles 2w,2w+1 of a
// 16-row stripe.  KT is a compile-time constant so the 1-deep software
// pipeline fully unrolls: straight-line 6 loads + 2 WMMAs per k-stage, two
// independent accumulator chains, no per-iteration branches.
// Optional bf16 mirror of the output (Cbf) feeds the next GEMM in the chain.
// ---------------------------------------------------------------------------
template <int KT>
__global__ __launch_bounds__(256) void gemm_bf16_n256(
    const unsigned short* __restrict__ A, int lda,
    const unsigned short* __restrict__ Bp,
    const float* __restrict__ bias,
    float* __restrict__ C, unsigned short* __restrict__ Cbf,
    int accumulate) {
  const int lane   = threadIdx.x & 31;
  const int wave   = threadIdx.x >> 5;
  const int laneLo = lane & 15;
  const int laneHi = lane >> 4;
  const int mbase  = blockIdx.x * 16;
  const unsigned short* Arow = A + (size_t)(mbase + laneLo) * lda;
  const unsigned short* bp   = Bp + (size_t)(wave * 2) * 512 + (size_t)lane * 16;

  v8f acc0 = {}; v8f acc1 = {};

  // -- prologue: load tile kt=0 --
  const int kc0 = laneHi * 8;
  v4u a0 = *(const v4u*)(Arow + kc0);
  v4u a1 = *(const v4u*)(Arow + kc0 + 16);
  v8u b0 = *(const v8u*)(bp);
  v8u b1 = *(const v8u*)(bp + 512);

  #pragma unroll
  for (int kt = 0; kt < KT; ++kt) {
    v4u na0 = {}, na1 = {};
    v8u nb0 = {}, nb1 = {};
    if (kt + 1 < KT) {                      // folds at compile time
      const int nkc = (kt + 1) * 32 + laneHi * 8;
      if (kt + 2 < KT) __builtin_prefetch(Arow + nkc + 32, 0, 1);
      na0 = *(const v4u*)(Arow + nkc);
      na1 = *(const v4u*)(Arow + nkc + 16);
      const unsigned short* nbp = bp + (size_t)(kt + 1) * (16 * 512);
      nb0 = *(const v8u*)(nbp);
      nb1 = *(const v8u*)(nbp + 512);
    }
    v8u au;
    #pragma unroll
    for (int q = 0; q < 4; ++q) { au[q] = a0[q]; au[4 + q] = a1[q]; }
    v16bf av  = __builtin_bit_cast(v16bf, au);
    v16bf bv0 = __builtin_bit_cast(v16bf, b0);
    v16bf bv1 = __builtin_bit_cast(v16bf, b1);
    acc0 = __builtin_amdgcn_wmma_f32_16x16x32_bf16(false, av, false, bv0,
                                                   (short)0, acc0, false, false);
    acc1 = __builtin_amdgcn_wmma_f32_16x16x32_bf16(false, av, false, bv1,
                                                   (short)0, acc1, false, false);
    a0 = na0; a1 = na1; b0 = nb0; b1 = nb1;
  }

  const int n0 = wave * 32 + laneLo;
  const float b0v = bias ? bias[n0] : 0.f;
  const float b1v = bias ? bias[n0 + 16] : 0.f;
  #pragma unroll
  for (int r = 0; r < 8; ++r) {
    const size_t i0 = (size_t)(mbase + laneHi * 8 + r) * Hd + n0;
    float v0 = acc0[r] + b0v;
    float v1 = acc1[r] + b1v;
    if (accumulate) { v0 += C[i0]; v1 += C[i0 + 16]; }
    if (C)   { C[i0] = v0;          C[i0 + 16] = v1; }
    if (Cbf) { Cbf[i0] = f2bf(v0);  Cbf[i0 + 16] = f2bf(v1); }
  }
}

// c = coord @ W_coord + b_coord   (K=3, pure VALU) -> bf16 directly
__global__ __launch_bounds__(256) void coordc_kernel(
    const float* __restrict__ coord, const float* __restrict__ Wc,
    const float* __restrict__ bc, unsigned short* __restrict__ out, int total) {
  int i = blockIdx.x * 256 + threadIdx.x;
  if (i >= total) return;
  int n = i >> 8, j = i & (Hd - 1);
  out[i] = f2bf(bc[j] + coord[n * 3] * Wc[j] + coord[n * 3 + 1] * Wc[Hd + j]
                      + coord[n * 3 + 2] * Wc[2 * Hd + j]);
}

// ve[j] = W_edge . We[:,j] ;  ce[j] = b_edge . We[:,j] + b_agg[l][j]
__global__ __launch_bounds__(256) void vece_kernel(
    const float* __restrict__ We, const float* __restrict__ Wedge,
    const float* __restrict__ bedge, const float* __restrict__ bagg,
    float* __restrict__ ve, float* __restrict__ ce) {
  int j = blockIdx.x * 256 + threadIdx.x;
  if (j < Hd) {
    float s = 0.f;
    for (int k = 0; k < Hd; ++k) s += Wedge[k] * We[(size_t)k * Hd + j];
    ve[j] = s;
  } else if (j < 2 * Hd) {
    int jj = j - Hd;
    float s = bagg[jj];
    for (int k = 0; k < Hd; ++k) s += bedge[k] * We[(size_t)k * Hd + jj];
    ce[jj] = s;
  }
}

// One wave per edge; lane owns 8 contiguous channels. agg += r^exp * (...)
__global__ __launch_bounds__(256) void edge_kernel(
    const float* __restrict__ hs, const float* __restrict__ hd,
    const int* __restrict__ src, const int* __restrict__ dst,
    const float* __restrict__ radius, const float* __restrict__ ve,
    const float* __restrict__ ce, const float* __restrict__ exp_l, int l,
    float* __restrict__ agg, int E) {
  int lane = threadIdx.x & 31;
  int e = blockIdx.x * 8 + (threadIdx.x >> 5);
  if (e >= E) return;
  int s = src[e], d = dst[e];
  float r = radius[e];
  float w = __expf(__logf(r) * exp_l[l]);     // radius^exp_l
  int c = lane * 8;
  const float4* ps = (const float4*)(hs + (size_t)s * Hd + c);
  const float4* pd = (const float4*)(hd + (size_t)d * Hd + c);
  const float4* pv = (const float4*)(ve + c);
  const float4* pc = (const float4*)(ce + c);
  float4 a0 = ps[0], a1 = ps[1], b0 = pd[0], b1 = pd[1];
  float4 v0 = pv[0], v1 = pv[1], e0 = pc[0], e1 = pc[1];
  float* out = agg + (size_t)d * Hd + c;
  float m[8];
  m[0] = w * (a0.x + b0.x + r * v0.x + e0.x);
  m[1] = w * (a0.y + b0.y + r * v0.y + e0.y);
  m[2] = w * (a0.z + b0.z + r * v0.z + e0.z);
  m[3] = w * (a0.w + b0.w + r * v0.w + e0.w);
  m[4] = w * (a1.x + b1.x + r * v1.x + e1.x);
  m[5] = w * (a1.y + b1.y + r * v1.y + e1.y);
  m[6] = w * (a1.z + b1.z + r * v1.z + e1.z);
  m[7] = w * (a1.w + b1.w + r * v1.w + e1.w);
  #pragma unroll
  for (int q = 0; q < 8; ++q)
    __hip_atomic_fetch_add(out + q, m[q], __ATOMIC_RELAXED,
                           __HIP_MEMORY_SCOPE_AGENT);
}

// rst = (1+eps)*h + agg  -> bf16 (feeds GLU GEMMs)
__global__ __launch_bounds__(256) void rst_kernel(
    const float* __restrict__ h, const float* __restrict__ agg,
    unsigned short* __restrict__ rstbf,
    const float* __restrict__ eps_l, int l, int total) {
  int i = blockIdx.x * 256 + threadIdx.x;
  if (i < total) rstbf[i] = f2bf((1.f + eps_l[l]) * h[i] + agg[i]);
}

// h <- ga * sigmoid(gb)   (fp32 for readout + bf16 for next-layer GEMMs)
__global__ __launch_bounds__(256) void glu_kernel(
    const float* __restrict__ ga, const float* __restrict__ gb,
    float* __restrict__ h, unsigned short* __restrict__ hbf, int total) {
  int i = blockIdx.x * 256 + threadIdx.x;
  if (i < total) {
    float v = ga[i] * (1.f / (1.f + __expf(-gb[i])));
    h[i] = v; hbf[i] = f2bf(v);
  }
}

// masked segment-sum into feat[16 x 256]
__global__ __launch_bounds__(256) void reduce_kernel(
    const float* __restrict__ h, const int* __restrict__ mask,
    const int* __restrict__ gid, float* __restrict__ feat, int total) {
  int i = blockIdx.x * 256 + threadIdx.x;
  if (i >= total) return;
  int n = i >> 8, j = i & (Hd - 1);
  if (mask[n] != 0)
    __hip_atomic_fetch_add(&feat[(size_t)gid[n] * Hd + j], h[i],
                           __ATOMIC_RELAXED, __HIP_MEMORY_SCOPE_AGENT);
}

// out = sigmoid(feat/N @ W_mlp + b_mlp)   (16x100, tiny)
__global__ __launch_bounds__(256) void mlp_kernel(
    const float* __restrict__ feat, const float* __restrict__ Wm,
    const float* __restrict__ bm, float* __restrict__ out, float invN) {
  int i = blockIdx.x * 256 + threadIdx.x;
  if (i >= Gg * OD) return;
  int g = i / OD, j = i - g * OD;
  float s = bm[j];
  for (int k = 0; k < Hd; ++k)
    s += feat[(size_t)g * Hd + k] * invN * Wm[(size_t)k * OD + j];
  out[i] = 1.f / (1.f + __expf(-s));
}

extern "C" void kernel_launch(void* const* d_in, const int* in_sizes, int n_in,
                              void* d_out, int out_size, void* d_ws, size_t ws_size,
                              hipStream_t stream) {
  const float* atomic_num = (const float*)d_in[0];
  const float* coord      = (const float*)d_in[1];
  const float* radius     = (const float*)d_in[2];
  const int*   src        = (const int*)d_in[3];
  const int*   dst        = (const int*)d_in[4];
  const int*   abs_mask   = (const int*)d_in[5];
  const int*   graph_id   = (const int*)d_in[6];
  const float* W_atom     = (const float*)d_in[7];
  const float* b_atom     = (const float*)d_in[8];
  const float* W_coord    = (const float*)d_in[9];
  const float* b_coord    = (const float*)d_in[10];
  const float* W_node     = (const float*)d_in[11];
  const float* b_node     = (const float*)d_in[12];
  const float* W_edge     = (const float*)d_in[13];
  const float* b_edge     = (const float*)d_in[14];
  const float* W_agg      = (const float*)d_in[15];
  const float* b_agg      = (const float*)d_in[16];
  const float* W_glu_a    = (const float*)d_in[17];
  const float* b_glu_a    = (const float*)d_in[18];
  const float* W_glu_b    = (const float*)d_in[19];
  const float* b_glu_b    = (const float*)d_in[20];
  const float* exp_l      = (const float*)d_in[21];
  const float* eps_l      = (const float*)d_in[22];
  const float* W_mlp      = (const float*)d_in[23];
  const float* b_mlp      = (const float*)d_in[24];
  float* out = (float*)d_out;

  const size_t NH = (size_t)Nn * Hd;
  float* h    = (float*)d_ws;
  float* t0   = h   + NH;          // hs / GLU-a scratch (fp32)
  float* t1   = t0  + NH;          // hd / GLU-b scratch (fp32)
  float* agg  = t1  + NH;          // agg (fp32 atomics)
  unsigned short* hbf   = (unsigned short*)(agg + NH);   // bf16 h
  unsigned short* rstbf = hbf + NH;                      // bf16 rst
  float* ve   = (float*)(rstbf + NH);
  float* ce   = ve + Hd;
  float* feat = ce + Hd;
  unsigned short* pWatom = (unsigned short*)(feat + (size_t)Gg * Hd); // KT=4
  unsigned short* pWnTop = pWatom + 4 * 16 * 512;                     // KT=8
  unsigned short* pWnBot = pWnTop + 8 * 16 * 512;
  unsigned short* pWs    = pWnBot + 8 * 16 * 512;
  unsigned short* pWd    = pWs    + 8 * 16 * 512;
  unsigned short* pGa    = pWd    + 8 * 16 * 512;
  unsigned short* pGb    = pGa    + 8 * 16 * 512;
  // bf16 aliases of dead fp32 regions during the embedding phase
  unsigned short* abf  = rstbf;               // padded atomic_num (N x 128)
  unsigned short* t0bf = (unsigned short*)t0; // a  (N x 256 bf16)
  unsigned short* t1bf = (unsigned short*)t1; // c  (N x 256 bf16)

  const int KT128 = ADP / 32;             // 4
  const int KT256 = Hd / 32;              // 8
  const int gM    = Nn / 16;              // 1250 row stripes (M % 16 == 0)
  const int gElem = ((int)NH + 255) / 256;

  // ---- node embedding: h = [a|c] @ W_node + b  (split-K, no concat) ----
  pack_b_kernel<<<(KT128 * 8192 + 255) / 256, 256, 0, stream>>>(W_atom, AD, Hd, pWatom, KT128);
  pack_b_kernel<<<(KT256 * 8192 + 255) / 256, 256, 0, stream>>>(W_node, Hd, Hd, pWnTop, KT256);
  pack_b_kernel<<<(KT256 * 8192 + 255) / 256, 256, 0, stream>>>(W_node + (size_t)Hd * Hd, Hd, Hd, pWnBot, KT256);
  cvt_atom_kernel<<<(Nn * ADP + 255) / 256, 256, 0, stream>>>(atomic_num, abf, Nn * ADP);
  gemm_bf16_n256<4><<<gM, 256, 0, stream>>>(abf, ADP, pWatom, b_atom, nullptr, t0bf, 0);
  coordc_kernel<<<gElem, 256, 0, stream>>>(coord, W_coord, b_coord, t1bf, (int)NH);
  gemm_bf16_n256<8><<<gM, 256, 0, stream>>>(t0bf, Hd, pWnTop, b_node, h, nullptr, 0);
  gemm_bf16_n256<8><<<gM, 256, 0, stream>>>(t1bf, Hd, pWnBot, nullptr, h, hbf, 1);

  // ---- message-passing layers ----
  for (int l = 0; l < Ll; ++l) {
    const float* Ws = W_agg + (size_t)(l * 3 * Hd) * Hd;
    const float* We = Ws + (size_t)Hd * Hd;
    const float* Wd = We + (size_t)Hd * Hd;
    pack_b_kernel<<<(KT256 * 8192 + 255) / 256, 256, 0, stream>>>(Ws, Hd, Hd, pWs, KT256);
    pack_b_kernel<<<(KT256 * 8192 + 255) / 256, 256, 0, stream>>>(Wd, Hd, Hd, pWd, KT256);
    pack_b_kernel<<<(KT256 * 8192 + 255) / 256, 256, 0, stream>>>(W_glu_a + (size_t)l * Hd * Hd, Hd, Hd, pGa, KT256);
    pack_b_kernel<<<(KT256 * 8192 + 255) / 256, 256, 0, stream>>>(W_glu_b + (size_t)l * Hd * Hd, Hd, Hd, pGb, KT256);
    vece_kernel<<<2, 256, 0, stream>>>(We, W_edge, b_edge, b_agg + l * Hd, ve, ce);

    gemm_bf16_n256<8><<<gM, 256, 0, stream>>>(hbf, Hd, pWs, nullptr, t0, nullptr, 0);
    gemm_bf16_n256<8><<<gM, 256, 0, stream>>>(hbf, Hd, pWd, nullptr, t1, nullptr, 0);

    hipMemsetAsync(agg, 0, NH * sizeof(float), stream);
    edge_kernel<<<(Ee + 7) / 8, 256, 0, stream>>>(t0, t1, src, dst, radius, ve, ce, exp_l, l, agg, Ee);
    rst_kernel<<<gElem, 256, 0, stream>>>(h, agg, rstbf, eps_l, l, (int)NH);

    gemm_bf16_n256<8><<<gM, 256, 0, stream>>>(rstbf, Hd, pGa, b_glu_a + l * Hd, t0, nullptr, 0);
    gemm_bf16_n256<8><<<gM, 256, 0, stream>>>(rstbf, Hd, pGb, b_glu_b + l * Hd, t1, nullptr, 0);
    glu_kernel<<<gElem, 256, 0, stream>>>(t0, t1, h, hbf, (int)NH);
  }

  // ---- readout ----
  hipMemsetAsync(feat, 0, (size_t)Gg * Hd * sizeof(float), stream);
  reduce_kernel<<<gElem, 256, 0, stream>>>(h, abs_mask, graph_id, feat, (int)NH);
  mlp_kernel<<<(Gg * OD + 255) / 256, 256, 0, stream>>>(feat, W_mlp, b_mlp, out, 1.0f / (float)Nn);
}
